// EnhancedGNNCapModel_37168646979918
// MI455X (gfx1250) — matrix operand order repaired and checked by reading, not compiled, not run on hardware
//
#include <hip/hip_runtime.h>
#include <hip/hip_bf16.h>

// ---------------- types ----------------
typedef __bf16 v16bf __attribute__((ext_vector_type(16)));
typedef float  v8f   __attribute__((ext_vector_type(8)));
typedef unsigned int u32x4 __attribute__((ext_vector_type(4)));
typedef unsigned int u32x2 __attribute__((ext_vector_type(2)));
typedef float  f32x4 __attribute__((ext_vector_type(4)));

union AFrag { u32x4 q[2]; v16bf bf; };   // 8 VGPRs = one 16x32 (or 32x16) bf16 fragment

#define N_NODES 50000
#define N_EDGES 600000
#define IN_CH   128
#define K1PAD   288          // 263 padded to 9*32
#define NKT1    9
#define NKT2    4

__device__ __forceinline__ unsigned short f2bf(float f) {
    unsigned int u = __float_as_uint(f);
    unsigned int r = u + 0x7FFFu + ((u >> 16) & 1u);   // RNE
    return (unsigned short)(r >> 16);
}

// ---------------- pre-pass: zero / convert ----------------
__global__ void zero_f32(float* __restrict__ p, int n4) {
    int t = blockIdx.x * blockDim.x + threadIdx.x;
    if (t < n4) ((f32x4*)p)[t] = (f32x4){0.f, 0.f, 0.f, 0.f};
}

__global__ void cvt_bf16(const float* __restrict__ src, unsigned short* __restrict__ dst, int n4) {
    int t = blockIdx.x * blockDim.x + threadIdx.x;
    if (t < n4) {
        f32x4 f = ((const f32x4*)src)[t];
        u32x2 o;
        o.x = (unsigned)f2bf(f.x) | ((unsigned)f2bf(f.y) << 16);
        o.y = (unsigned)f2bf(f.z) | ((unsigned)f2bf(f.w) << 16);
        ((u32x2*)dst)[t] = o;
    }
}

// ---------------- weight repack to B-fragment-native layout ----------------
// Block b = kt*NT + nt holds the 32x16 (KxN) bf16 tile; within a block each lane
// owns 8 contiguous dwords (32B): dword v = bf16 pair (K0, K0+1) of column n,
//   n  = nt*16 + (lane&15),   K0 = kt*32 + (lane>>4)*16 + 2*v
// so the kernel-side B fragment load is two b128 loads per lane.
__global__ void pack_w1(const float* __restrict__ W1, unsigned int* __restrict__ dst) {
    int t = blockIdx.x * blockDim.x + threadIdx.x;          // dword id
    if (t >= NKT1 * 8 * 256) return;
    int v = t & 7, lane = (t >> 3) & 31, blk = t >> 8;
    int nt = blk & 7, kt = blk >> 3;
    int n  = nt * 16 + (lane & 15);
    int K0 = kt * 32 + ((lane >> 4) << 4) + 2 * v;
    float lo = (K0     < 263) ? W1[K0       * 128 + n] : 0.f;
    float hi = (K0 + 1 < 263) ? W1[(K0 + 1) * 128 + n] : 0.f;
    dst[t] = (unsigned)f2bf(lo) | ((unsigned)f2bf(hi) << 16);
}

__global__ void pack_w2(const float* __restrict__ W2, unsigned int* __restrict__ dst) {
    int t = blockIdx.x * blockDim.x + threadIdx.x;
    if (t >= NKT2 * 8 * 256) return;
    int v = t & 7, lane = (t >> 3) & 31, blk = t >> 8;
    int nt = blk & 7, kt = blk >> 3;
    int n  = nt * 16 + (lane & 15);
    int K0 = kt * 32 + ((lane >> 4) << 4) + 2 * v;
    float lo = W2[K0 * 128 + n], hi = W2[(K0 + 1) * 128 + n];
    dst[t] = (unsigned)f2bf(lo) | ((unsigned)f2bf(hi) << 16);
}

// B_x[128 x 512]: cols 0..127 = Wg[0:128]+Wg[256:384] (gate, x part); 128..511 = W_hh^T
__global__ void pack_bx(const float* __restrict__ Wg, const float* __restrict__ Whh,
                        unsigned int* __restrict__ dst) {
    int t = blockIdx.x * blockDim.x + threadIdx.x;
    if (t >= NKT2 * 32 * 256) return;
    int v = t & 7, lane = (t >> 3) & 31, blk = t >> 8;
    int nt = blk & 31, kt = blk >> 5;
    int n  = nt * 16 + (lane & 15);
    int K0 = kt * 32 + ((lane >> 4) << 4) + 2 * v;
    float lo, hi;
    if (n < 128) { lo = Wg[K0*128+n] + Wg[(256+K0)*128+n];  hi = Wg[(K0+1)*128+n] + Wg[(257+K0)*128+n]; }
    else         { int j = n - 128;  lo = Whh[j*128 + K0];  hi = Whh[j*128 + K0 + 1]; }
    dst[t] = (unsigned)f2bf(lo) | ((unsigned)f2bf(hi) << 16);
}

// B_a[128 x 512]: cols 0..127 = Wg[128:256] (gate, aggr part); 128..511 = W_ih^T
__global__ void pack_ba(const float* __restrict__ Wg, const float* __restrict__ Wih,
                        unsigned int* __restrict__ dst) {
    int t = blockIdx.x * blockDim.x + threadIdx.x;
    if (t >= NKT2 * 32 * 256) return;
    int v = t & 7, lane = (t >> 3) & 31, blk = t >> 8;
    int nt = blk & 31, kt = blk >> 5;
    int n  = nt * 16 + (lane & 15);
    int K0 = kt * 32 + ((lane >> 4) << 4) + 2 * v;
    float lo, hi;
    if (n < 128) { lo = Wg[(128+K0)*128+n];  hi = Wg[(129+K0)*128+n]; }
    else         { int j = n - 128;  lo = Wih[j*128 + K0];  hi = Wih[j*128 + K0 + 1]; }
    dst[t] = (unsigned)f2bf(lo) | ((unsigned)f2bf(hi) << 16);
}

// ---------------- edge kernel: gather -> 2-layer MLP (WMMA) -> atomic scatter ----------------
__global__ __launch_bounds__(256) void edge_kernel(
    const unsigned short* __restrict__ xbf,     // [N,128] bf16
    const int*            __restrict__ ei,      // [2,E]
    const float*          __restrict__ eattr,   // [E,7]
    const float*          __restrict__ b1,
    const float*          __restrict__ b2,
    const unsigned short* __restrict__ W1p,     // packed fragments
    const unsigned short* __restrict__ W2p,
    float*                __restrict__ aggr)    // [N,128] f32, atomic dest
{
    __shared__ __align__(16) unsigned short msg[128 * K1PAD];   // 72 KB
    __shared__ __align__(16) unsigned short hid[128 * 128];     // 32 KB
    __shared__ int dst_s[128];

    const int tid = threadIdx.x;
    const int wgBase = blockIdx.x * 128;

    // ---- gather: 2 threads per edge row ----
    {
        int r = tid >> 1, h = tid & 1;
        int e = wgBase + r;
        if (e < N_EDGES) {
            int s = ei[e];
            int d = ei[N_EDGES + e];
            if (h == 0) dst_s[r] = d;
            const u32x4* xi = (const u32x4*)(xbf + d * 128 + h * 64);
            const u32x4* xj = (const u32x4*)(xbf + s * 128 + h * 64);
            u32x4* mi = (u32x4*)(msg + r * K1PAD + h * 64);
            u32x4* mj = (u32x4*)(msg + r * K1PAD + 128 + h * 64);
#pragma unroll
            for (int q = 0; q < 8; ++q) { mi[q] = xi[q]; mj[q] = xj[q]; }
            // attr cols 256..262, zero-pad 263..287 (padded W1 rows are zero too)
            int cbase = 256 + h * 16;
#pragma unroll
            for (int c = 0; c < 16; ++c) {
                int col = cbase + c, j = col - 256;
                float v = (j < 7) ? eattr[e * 7 + j] : 0.f;
                msg[r * K1PAD + col] = f2bf(v);
            }
        }
    }
    __syncthreads();

    const int wv = tid >> 5, lane = tid & 31;
    const int r0 = wv * 16;
    if (wgBase + r0 >= N_EDGES) return;           // whole-wave uniform exit, EXEC stays full
    const int m = lane & 15;
    const int hi16 = lane >> 4;
    const int kh8 = hi16 << 4;

    // ---- stage 1: [16x288] @ W1p -> ReLU -> hid bf16 ----
#pragma unroll 1
    for (int nt = 0; nt < 8; ++nt) {
        float bv = b1[nt * 16 + m];
        v8f c = {bv, bv, bv, bv, bv, bv, bv, bv};
#pragma unroll
        for (int kt = 0; kt < NKT1; ++kt) {
            AFrag a, b;
            const unsigned short* ap = msg + (r0 + m) * K1PAD + kt * 32 + kh8;
            a.q[0] = *(const u32x4*)ap;
            a.q[1] = *(const u32x4*)(ap + 8);
            const u32x4* bp = (const u32x4*)(W1p + (kt * 8 + nt) * 512 + lane * 16);
            b.q[0] = bp[0]; b.q[1] = bp[1];
            c = __builtin_amdgcn_wmma_f32_16x16x32_bf16(false, a.bf, false, b.bf, (short)0, c, false, false);
        }
#pragma unroll
        for (int v = 0; v < 8; ++v) {
            int M = v + (hi16 << 3);
            float rl = c[v] > 0.f ? c[v] : 0.f;
            hid[(r0 + M) * 128 + nt * 16 + m] = f2bf(rl);
        }
    }

    // ---- stage 2: hid @ W2p -> atomic scatter-add into aggr[dst] ----
#pragma unroll 1
    for (int nt = 0; nt < 8; ++nt) {
        float bv = b2[nt * 16 + m];
        v8f c = {bv, bv, bv, bv, bv, bv, bv, bv};
#pragma unroll
        for (int kt = 0; kt < NKT2; ++kt) {
            AFrag a, b;
            const unsigned short* ap = hid + (r0 + m) * 128 + kt * 32 + kh8;
            a.q[0] = *(const u32x4*)ap;
            a.q[1] = *(const u32x4*)(ap + 8);
            const u32x4* bp = (const u32x4*)(W2p + (kt * 8 + nt) * 512 + lane * 16);
            b.q[0] = bp[0]; b.q[1] = bp[1];
            c = __builtin_amdgcn_wmma_f32_16x16x32_bf16(false, a.bf, false, b.bf, (short)0, c, false, false);
        }
#pragma unroll
        for (int v = 0; v < 8; ++v) {
            int M = v + (hi16 << 3);
            int d = dst_s[r0 + M];
            atomicAdd(&aggr[d * 128 + nt * 16 + m], c[v]);   // global_atomic_add_f32, L2-resident
        }
    }
}

// ---------------- node kernel: gate + GRU GEMMs (WMMA) + elementwise + LayerNorm ----------------
__global__ __launch_bounds__(256) void node_kernel(
    const float*          __restrict__ x,      // [N,128] f32
    const unsigned short* __restrict__ xbf,    // [N,128] bf16
    const float*          __restrict__ aggr,   // [N,128] f32
    const unsigned short* __restrict__ Bxp,    // packed [x-side] B: 128x512
    const unsigned short* __restrict__ Bap,    // packed [aggr-side] B: 128x512
    const float* __restrict__ bg,
    const float* __restrict__ b_ih,
    const float* __restrict__ b_hh,
    const float* __restrict__ gamma,
    const float* __restrict__ beta,
    float* __restrict__ out)
{
    __shared__ __align__(16) unsigned short xs[16 * 128];
    __shared__ __align__(16) unsigned short as_[16 * 128];
    __shared__ __align__(16) float Sx[16 * 512];   // [gate_x | h_r | h_z | h_n]
    __shared__ __align__(16) float Sa[16 * 512];   // [gate_a | i_r | i_z | i_n]

    const int tid = threadIdx.x;
    const int nodeBase = blockIdx.x * 16;

    // x tile (already bf16): 4 KB, one b128 per thread
    ((u32x4*)xs)[tid] = ((const u32x4*)(xbf + nodeBase * 128))[tid];
    // aggr tile: f32 -> bf16
    {
        const f32x4* ap = (const f32x4*)(aggr + nodeBase * 128);
        f32x4 f0 = ap[tid * 2], f1 = ap[tid * 2 + 1];
        u32x4 o;
        o.x = (unsigned)f2bf(f0.x) | ((unsigned)f2bf(f0.y) << 16);
        o.y = (unsigned)f2bf(f0.z) | ((unsigned)f2bf(f0.w) << 16);
        o.z = (unsigned)f2bf(f1.x) | ((unsigned)f2bf(f1.y) << 16);
        o.w = (unsigned)f2bf(f1.z) | ((unsigned)f2bf(f1.w) << 16);
        ((u32x4*)as_)[tid] = o;
    }
    __syncthreads();

    const int wv = tid >> 5, lane = tid & 31;
    const int m = lane & 15, hi16 = lane >> 4, kh8 = hi16 << 4;

    // each wave owns 4 of the 32 n-tiles for BOTH GEMMs
#pragma unroll 1
    for (int i = 0; i < 4; ++i) {
        int ntg = wv * 4 + i;
        v8f cx = {0.f, 0.f, 0.f, 0.f, 0.f, 0.f, 0.f, 0.f};
        v8f ca = {0.f, 0.f, 0.f, 0.f, 0.f, 0.f, 0.f, 0.f};
#pragma unroll
        for (int kt = 0; kt < NKT2; ++kt) {
            AFrag ax, aa, bx, ba;
            const unsigned short* axp = xs  + m * 128 + kt * 32 + kh8;
            const unsigned short* aap = as_ + m * 128 + kt * 32 + kh8;
            ax.q[0] = *(const u32x4*)axp;  ax.q[1] = *(const u32x4*)(axp + 8);
            aa.q[0] = *(const u32x4*)aap;  aa.q[1] = *(const u32x4*)(aap + 8);
            const u32x4* bxp = (const u32x4*)(Bxp + (kt * 32 + ntg) * 512 + lane * 16);
            const u32x4* bap = (const u32x4*)(Bap + (kt * 32 + ntg) * 512 + lane * 16);
            bx.q[0] = bxp[0]; bx.q[1] = bxp[1];
            ba.q[0] = bap[0]; ba.q[1] = bap[1];
            cx = __builtin_amdgcn_wmma_f32_16x16x32_bf16(false, ax.bf, false, bx.bf, (short)0, cx, false, false);
            ca = __builtin_amdgcn_wmma_f32_16x16x32_bf16(false, aa.bf, false, ba.bf, (short)0, ca, false, false);
        }
#pragma unroll
        for (int v = 0; v < 8; ++v) {
            int M = v + (hi16 << 3);
            Sx[M * 512 + ntg * 16 + m] = cx[v];
            Sa[M * 512 + ntg * 16 + m] = ca[v];
        }
    }
    __syncthreads();

    // elementwise GRU + gate blend + LayerNorm: 16 threads per row, 8 ch each
    int row = tid >> 4, sub = tid & 15;
    int node = nodeBase + row;
    const float* xr = x + node * 128;
    float o[8], sum = 0.f, sq = 0.f;
#pragma unroll
    for (int q = 0; q < 8; ++q) {
        int c = sub * 8 + q;
        float xv  = xr[c];
        float gr  = Sx[row * 512 + c] + Sa[row * 512 + c] + bg[c];
        float hr  = Sx[row * 512 + 128 + c] + b_hh[c];
        float hz  = Sx[row * 512 + 256 + c] + b_hh[128 + c];
        float hn  = Sx[row * 512 + 384 + c] + b_hh[256 + c];
        float ir  = Sa[row * 512 + 128 + c] + b_ih[c];
        float iz  = Sa[row * 512 + 256 + c] + b_ih[128 + c];
        float in_ = Sa[row * 512 + 384 + c] + b_ih[256 + c];
        float r = 1.f / (1.f + __expf(-(ir + hr)));
        float z = 1.f / (1.f + __expf(-(iz + hz)));
        float nn = tanhf(in_ + r * hn);
        float upd = (1.f - z) * nn + z * xv;
        float g = 1.f / (1.f + __expf(-gr));
        float ov = g * upd + (1.f - g) * xv;
        o[q] = ov;  sum += ov;  sq += ov * ov;
    }
    // 16-lane row reduction (wave32: lanes [0..15]/[16..31] = 2 rows per wave)
#pragma unroll
    for (int off = 1; off < 16; off <<= 1) {
        sum += __shfl_xor(sum, off, 16);
        sq  += __shfl_xor(sq,  off, 16);
    }
    float mean = sum * (1.f / 128.f);
    float var  = sq * (1.f / 128.f) - mean * mean;
    float inv  = rsqrtf(var + 1e-5f);
    f32x4 w0, w1;
    {
        int c = sub * 8;
        w0.x = (o[0] - mean) * inv * gamma[c+0] + beta[c+0];
        w0.y = (o[1] - mean) * inv * gamma[c+1] + beta[c+1];
        w0.z = (o[2] - mean) * inv * gamma[c+2] + beta[c+2];
        w0.w = (o[3] - mean) * inv * gamma[c+3] + beta[c+3];
        w1.x = (o[4] - mean) * inv * gamma[c+4] + beta[c+4];
        w1.y = (o[5] - mean) * inv * gamma[c+5] + beta[c+5];
        w1.z = (o[6] - mean) * inv * gamma[c+6] + beta[c+6];
        w1.w = (o[7] - mean) * inv * gamma[c+7] + beta[c+7];
        f32x4* op = (f32x4*)(out + node * 128 + c);
        op[0] = w0; op[1] = w1;
    }
}

// ---------------- launcher ----------------
extern "C" void kernel_launch(void* const* d_in, const int* in_sizes, int n_in,
                              void* d_out, int out_size, void* d_ws, size_t ws_size,
                              hipStream_t stream) {
    const float* x    = (const float*)d_in[0];
    const int*   ei   = (const int*)d_in[1];
    const float* ea   = (const float*)d_in[2];
    const float* W1   = (const float*)d_in[3];
    const float* b1   = (const float*)d_in[4];
    const float* W2   = (const float*)d_in[5];
    const float* b2   = (const float*)d_in[6];
    const float* Wg   = (const float*)d_in[7];
    const float* bg   = (const float*)d_in[8];
    const float* Wih  = (const float*)d_in[9];
    const float* bih  = (const float*)d_in[10];
    const float* Whh  = (const float*)d_in[11];
    const float* bhh  = (const float*)d_in[12];
    const float* gam  = (const float*)d_in[13];
    const float* bet  = (const float*)d_in[14];
    (void)in_sizes; (void)n_in; (void)out_size; (void)ws_size;

    char* ws = (char*)d_ws;
    size_t off = 0;
    float*          aggr = (float*)(ws + off);          off += (size_t)N_NODES * 128 * 4;  // 25.6 MB
    unsigned short* xbf  = (unsigned short*)(ws + off); off += (size_t)N_NODES * 128 * 2;  // 12.8 MB
    unsigned short* W1p  = (unsigned short*)(ws + off); off += (size_t)NKT1 * 8 * 1024;    // 72 KB
    unsigned short* W2p  = (unsigned short*)(ws + off); off += (size_t)NKT2 * 8 * 1024;    // 32 KB
    unsigned short* Bxp  = (unsigned short*)(ws + off); off += (size_t)NKT2 * 32 * 1024;   // 128 KB
    unsigned short* Bap  = (unsigned short*)(ws + off); off += (size_t)NKT2 * 32 * 1024;   // 128 KB

    zero_f32<<<6250, 256, 0, stream>>>(aggr, N_NODES * 128 / 4);
    cvt_bf16<<<6250, 256, 0, stream>>>(x, xbf, N_NODES * 128 / 4);
    pack_w1<<<72,  256, 0, stream>>>(W1, (unsigned int*)W1p);
    pack_w2<<<32,  256, 0, stream>>>(W2, (unsigned int*)W2p);
    pack_bx<<<128, 256, 0, stream>>>(Wg, Whh, (unsigned int*)Bxp);
    pack_ba<<<128, 256, 0, stream>>>(Wg, Wih, (unsigned int*)Bap);

    edge_kernel<<<(N_EDGES + 127) / 128, 256, 0, stream>>>(xbf, ei, ea, b1, b2, W1p, W2p, aggr);
    node_kernel<<<N_NODES / 16, 256, 0, stream>>>(x, xbf, aggr, Bxp, Bap,
                                                  bg, bih, bhh, gam, bet, (float*)d_out);
}
